// TransformerSentRelFusion_37168646979700
// MI455X (gfx1250) — matrix operand order repaired and checked by reading, not compiled, not run on hardware
//
#include <hip/hip_runtime.h>

// ---------------------------------------------------------------------------
// CDNA5 (gfx1250) wave32 WMMA implementation of the TransformerSentRelFusion
// forward pass.  All heavy GEMMs use v_wmma_f32_16x16x32_bf16 (fp32 accum).
// The generic GEMM now uses GLOBAL_LOAD_ASYNC_TO_LDS_B128 (ASYNCcnt) with
// double-buffered LDS so global->LDS copies overlap the WMMA pipe.
// pe4 sinusoidal tensor ([128,128,3072] = 192MB fp32) is NEVER materialized:
// it is synthesized per-tile in LDS inside the pos_fusion GEMM so trig VALU
// work co-executes with the XDL WMMA pipe.
// ---------------------------------------------------------------------------

typedef __bf16 bf16_t;
typedef __attribute__((ext_vector_type(16))) __bf16 v16bf;
typedef __attribute__((ext_vector_type(8)))  __bf16 v8bf;
typedef __attribute__((ext_vector_type(8)))  float  v8f;

#define DEV static __device__ __forceinline__

DEV v16bf mk16(v8bf lo, v8bf hi) {
  v16bf r;
#pragma unroll
  for (int e = 0; e < 8; ++e) { r[e] = lo[e]; r[e + 8] = hi[e]; }
  return r;
}

DEV v8f wmma_bf16(v16bf a, v16bf b, v8f c) {
  // 8-arg form: (neg_a, A, neg_b, B, c_mod, C, reuse_a, reuse_b)
  return __builtin_amdgcn_wmma_f32_16x16x32_bf16(false, a, false, b, (short)0, c,
                                                 false, false);
}

DEV v8f zero8() {
  v8f z;
#pragma unroll
  for (int e = 0; e < 8; ++e) z[e] = 0.f;
  return z;
}

// LDS byte offset of a __shared__ object: flat-LDS aperture keeps the offset
// in addr[31:0] (ISA 10.2), so truncation of the generic pointer is the offset.
DEV unsigned lds_off(const void* p) { return (unsigned)(size_t)p; }

// ------------------------- problem constants -------------------------------
// B=32 S=64 INPUT=1024 H=768 NH=12 DH=64 L=128 FF=3072 E=300 V=20 C=3
#define BB   32
#define SS   64
#define INP  1024
#define HH   768
#define NHD  12
#define DHD  64
#define LL   128
#define FFD  3072
#define EE   300
#define EPAD 320
#define HQ   192      // H/4
#define NCLS 3
#define BLROWS (BB * LL)   // 4096
#define BSROWS (BB * SS)   // 2048
#define LLROWS (LL * LL)   // 16384

// ======================= elementwise / prep kernels ========================

__global__ void k_f32_to_bf16(const float* __restrict__ s, bf16_t* __restrict__ d,
                              unsigned int n) {
  unsigned int i = blockIdx.x * 256u + threadIdx.x;
  if (i < n) d[i] = (bf16_t)s[i];
}

__global__ void k_pad_rows_bf16(const float* __restrict__ s, bf16_t* __restrict__ d,
                                int srcRows, int dstRows, int cols) {
  int i = blockIdx.x * 256 + threadIdx.x;
  if (i >= dstRows * cols) return;
  int r = i / cols, c = i - r * cols;
  d[i] = (bf16_t)(r < srcRows ? s[(size_t)r * cols + c] : 0.f);
}

// A_rel[b*64+s][0:320) = rel_embed[ids[b,s]][0:300) zero padded to K=320
__global__ void k_gather_rel(const float* __restrict__ emb, const int* __restrict__ ids,
                             bf16_t* __restrict__ d) {
  int i = blockIdx.x * 256 + threadIdx.x;
  if (i >= BSROWS * EPAD) return;
  int row = i / EPAD, c = i - row * EPAD;
  float v = 0.f;
  if (c < EE) v = emb[(size_t)ids[row] * EE + c];
  d[i] = (bf16_t)v;
}

// x[b*128+l] = l<64 ? xsent[b*64+l] : xrel[b*64+l-64]; also bf16 copy
__global__ void k_interleave(const float* __restrict__ xs, const float* __restrict__ xr,
                             float* __restrict__ x, bf16_t* __restrict__ xb) {
  int i = blockIdx.x * 256 + threadIdx.x;
  if (i >= BLROWS * HH) return;
  int row = i / HH, c = i - row * HH;
  int b = row >> 7, l = row & 127;
  float v = (l < 64) ? xs[((size_t)(b * 64 + l)) * HH + c]
                     : xr[((size_t)(b * 64 + (l - 64))) * HH + c];
  x[i] = v;
  xb[i] = (bf16_t)v;
}

// q,k,v fp32 [b*L+l][H] -> head-major bf16 [b][h][l][d]; q gets +u / +vb
__global__ void k_qkv_prep(const float* __restrict__ qf, const float* __restrict__ kf,
                           const float* __restrict__ vf, const float* __restrict__ u,
                           const float* __restrict__ vb, bf16_t* __restrict__ qu,
                           bf16_t* __restrict__ qv, bf16_t* __restrict__ kb,
                           bf16_t* __restrict__ vh) {
  int i = blockIdx.x * 256 + threadIdx.x;
  if (i >= BB * NHD * LL * DHD) return;
  int d = i & 63;
  int l = (i >> 6) & 127;
  int rest = i >> 13;
  int h = rest % NHD, b = rest / NHD;
  size_t src = ((size_t)(b * LL + l)) * HH + h * DHD + d;
  int hd = h * DHD + d;
  float q = qf[src];
  qu[i] = (bf16_t)(q + u[hd]);
  qv[i] = (bf16_t)(q + vb[hd]);
  kb[i] = (bf16_t)kf[src];
  vh[i] = (bf16_t)vf[src];
}

// scale (1/sqrt(64)), key-mask, softmax over j (wave per row of 128)
__global__ __launch_bounds__(256) void k_softmax(const float* __restrict__ score,
                                                 bf16_t* __restrict__ attn,
                                                 const int* __restrict__ smask,
                                                 const int* __restrict__ rmask) {
  int wv = threadIdx.x >> 5, lane = threadIdx.x & 31;
  int row = blockIdx.x * 8 + wv;  // < 32*12*128
  int b = row / (NHD * LL);
  const float* sr = score + (size_t)row * LL;
  float v[4];
#pragma unroll
  for (int t = 0; t < 4; ++t) {
    int j = lane + t * 32;
    int m = (j < SS) ? smask[b * SS + j] : rmask[b * SS + (j - SS)];
    float x = sr[j] * 0.125f;
    v[t] = (m > 0) ? x : -1e9f;
  }
  float mx = fmaxf(fmaxf(v[0], v[1]), fmaxf(v[2], v[3]));
  for (int o = 16; o; o >>= 1) mx = fmaxf(mx, __shfl_xor(mx, o, 32));
  float s = 0.f;
#pragma unroll
  for (int t = 0; t < 4; ++t) { v[t] = __expf(v[t] - mx); s += v[t]; }
  for (int o = 16; o; o >>= 1) s += __shfl_xor(s, o, 32);
  float inv = 1.f / s;
  bf16_t* ar = attn + (size_t)row * LL;
#pragma unroll
  for (int t = 0; t < 4; ++t) ar[lane + t * 32] = (bf16_t)(v[t] * inv);
}

// out = LayerNorm(X + Y) * g + b (wave per 768-wide row)
__global__ __launch_bounds__(256) void k_add_ln(const float* __restrict__ X,
                                                const float* __restrict__ Y,
                                                const float* __restrict__ g,
                                                const float* __restrict__ b,
                                                float* __restrict__ oF,
                                                bf16_t* __restrict__ oB) {
  int wv = threadIdx.x >> 5, lane = threadIdx.x & 31;
  int row = blockIdx.x * 8 + wv;  // < 4096
  const float* xr = X + (size_t)row * HH;
  const float* yr = Y + (size_t)row * HH;
  float v[24];
  float s = 0.f;
#pragma unroll
  for (int i = 0; i < 24; ++i) { int c = lane + i * 32; v[i] = xr[c] + yr[c]; s += v[i]; }
  for (int o = 16; o; o >>= 1) s += __shfl_xor(s, o, 32);
  float mean = s * (1.f / HH);
  float vs = 0.f;
#pragma unroll
  for (int i = 0; i < 24; ++i) { float d = v[i] - mean; vs += d * d; }
  for (int o = 16; o; o >>= 1) vs += __shfl_xor(vs, o, 32);
  float inv = rsqrtf(vs * (1.f / HH) + 1e-5f);
#pragma unroll
  for (int i = 0; i < 24; ++i) {
    int c = lane + i * 32;
    float o = (v[i] - mean) * inv * g[c] + b[c];
    if (oF) oF[(size_t)row * HH + c] = o;
    if (oB) oB[(size_t)row * HH + c] = (bf16_t)o;
  }
}

// classifier head (K=192,C=3 tiny -> scalar), argmax + NLL per row
__global__ void k_cls(const float* __restrict__ fco, const float* __restrict__ cw,
                      const float* __restrict__ cb, const int* __restrict__ labels,
                      float* __restrict__ dout, float* __restrict__ nll,
                      float* __restrict__ valid) {
  int row = blockIdx.x * 256 + threadIdx.x;
  if (row >= BLROWS) return;
  float lg0 = cb[0], lg1 = cb[1], lg2 = cb[2];
  const float* xr = fco + (size_t)row * HQ;
  for (int k = 0; k < HQ; ++k) {
    float x = xr[k];
    lg0 += x * cw[k * 3 + 0];
    lg1 += x * cw[k * 3 + 1];
    lg2 += x * cw[k * 3 + 2];
  }
  int p = 0;
  float best = lg0;
  if (lg1 > best) { best = lg1; p = 1; }
  if (lg2 > best) { best = lg2; p = 2; }
  dout[1 + row] = (float)p;  // preds (second return value), cast to float
  float m = fmaxf(lg0, fmaxf(lg1, lg2));
  float lse = m + __logf(__expf(lg0 - m) + __expf(lg1 - m) + __expf(lg2 - m));
  int lab = labels[row];
  if (lab >= 0) {
    float pick = (lab == 0) ? lg0 : ((lab == 1) ? lg1 : lg2);
    nll[row] = lse - pick;
    valid[row] = 1.f;
  } else {
    nll[row] = 0.f;
    valid[row] = 0.f;
  }
}

__global__ __launch_bounds__(256) void k_loss(const float* __restrict__ nll,
                                              const float* __restrict__ valid,
                                              float* __restrict__ out) {
  __shared__ float sn[256], sv[256];
  float a = 0.f, b = 0.f;
  for (int i = threadIdx.x; i < BLROWS; i += 256) { a += nll[i]; b += valid[i]; }
  sn[threadIdx.x] = a;
  sv[threadIdx.x] = b;
  __syncthreads();
  for (int o = 128; o; o >>= 1) {
    if ((int)threadIdx.x < o) {
      sn[threadIdx.x] += sn[threadIdx.x + o];
      sv[threadIdx.x] += sv[threadIdx.x + o];
    }
    __syncthreads();
  }
  if (threadIdx.x == 0) out[0] = sn[0] / fmaxf(sv[0], 1.f);
}

// ========================== WMMA GEMM kernels ==============================
// Generic bf16 GEMM: C[M,N] = act(A[M,K] @ B[K,N] + bias).  128x128 block
// tile, 8 waves (4x2) of 32x64, K stepped by 32.  M%128==0, K%32==0, N%8==0.
// Global->LDS staging uses GLOBAL_LOAD_ASYNC_TO_LDS_B128 (ASYNCcnt) with
// double-buffered LDS: one barrier per K-step, async copy of tile k+1
// overlaps the WMMA math on tile k.  The 24-bit IOFFSET applies to BOTH the
// global and the LDS address, so chunk #2 is the same operands + offset:16.
__global__ __launch_bounds__(256) void k_gemm_bf16(
    const bf16_t* __restrict__ A, const bf16_t* __restrict__ B,
    const float* __restrict__ bias, float* __restrict__ Cf, bf16_t* __restrict__ Cb,
    int M, int N, int K, int relu) {
  __shared__ __align__(16) bf16_t As[2][128][40];   // +8 pad, 16B-mult row stride
  __shared__ __align__(16) bf16_t Bs[2][32][136];
  const int tid = threadIdx.x;
  const int wave = tid >> 5, lane = tid & 31;
  const int hsel = lane >> 4, l15 = lane & 15;
  const int m0 = blockIdx.y * 128, n0 = blockIdx.x * 128;
  const int wm = (wave & 3) * 32, wn = (wave >> 2) * 64;
  v8f acc[2][4];
#pragma unroll
  for (int i = 0; i < 2; ++i)
#pragma unroll
    for (int j = 0; j < 4; ++j) acc[i][j] = zero8();

  const int ar = tid >> 1, ac = (tid & 1) * 16;  // A staging: 128x32, 16 elems/thr
  const int br = tid >> 3, bc = (tid & 7) * 16;  // B staging: 32x128
  const bool bok = (n0 + bc < N);                // N%16==0 for guarded case (192)

  auto stage = [&](int buf, int kt) {
    {
      const bf16_t* s = A + (size_t)(m0 + ar) * K + kt + ac;
      unsigned d = lds_off(&As[buf][ar][ac]);
      asm volatile(
          "global_load_async_to_lds_b128 %0, %1, off\n\t"
          "global_load_async_to_lds_b128 %0, %1, off offset:16"
          :: "v"(d), "v"(s) : "memory");
    }
    if (bok) {
      const bf16_t* s = B + (size_t)(kt + br) * N + n0 + bc;
      unsigned d = lds_off(&Bs[buf][br][bc]);
      asm volatile(
          "global_load_async_to_lds_b128 %0, %1, off\n\t"
          "global_load_async_to_lds_b128 %0, %1, off offset:16"
          :: "v"(d), "v"(s) : "memory");
    } else {
#pragma unroll
      for (int e = 0; e < 16; ++e) Bs[buf][br][bc + e] = (bf16_t)0.f;
    }
  };

  stage(0, 0);  // prologue: async-copy tile 0
  for (int kt = 0; kt < K; kt += 32) {
    const int buf = (kt >> 5) & 1;
    asm volatile("s_wait_asynccnt 0" ::: "memory");  // my tile copies landed
    __syncthreads();                                 // everyone's landed
    if (kt + 32 < K) stage(buf ^ 1, kt + 32);        // overlap next tile copy
    if (kt + 64 < K) {                               // L2 prefetch 2 tiles out
      __builtin_prefetch(A + (size_t)(m0 + ar) * K + kt + 64 + ac, 0, 1);
      __builtin_prefetch(B + (size_t)(kt + 64 + br) * N + n0 + bc, 0, 1);
    }
    v16bf bfr[4];
#pragma unroll
    for (int nf = 0; nf < 4; ++nf) {  // B frag: lane%16 = N col, K split by half
#pragma unroll
      for (int s = 0; s < 16; ++s)
        bfr[nf][s] = Bs[buf][hsel * 8 + (s & 7) + (s >> 3) * 16][wn + nf * 16 + l15];
    }
#pragma unroll
    for (int mf = 0; mf < 2; ++mf) {  // A frag: lane%16 = M row, K split by half
      const int row = wm + mf * 16 + l15;
      v8bf a0 = *(const v8bf*)&As[buf][row][hsel * 8];
      v8bf a1 = *(const v8bf*)&As[buf][row][hsel * 8 + 16];
      v16bf af = mk16(a0, a1);
#pragma unroll
      for (int nf = 0; nf < 4; ++nf) acc[mf][nf] = wmma_bf16(af, bfr[nf], acc[mf][nf]);
    }
    // no trailing barrier needed: a wave signals the next barrier only after
    // its LDS reads completed (s_wait_dscnt precedes s_barrier_signal), and
    // the buffer it just read is only re-filled after that barrier.
  }
#pragma unroll
  for (int mf = 0; mf < 2; ++mf)
#pragma unroll
    for (int nf = 0; nf < 4; ++nf)
#pragma unroll
      for (int e = 0; e < 8; ++e) {
        int gr = m0 + wm + mf * 16 + e + 8 * hsel;
        int gc = n0 + wn + nf * 16 + l15;
        if (gc < N) {
          float v = acc[mf][nf][e];
          if (bias) v += bias[gc];
          if (relu) v = fmaxf(v, 0.f);
          size_t o = (size_t)gr * N + gc;
          if (Cf) Cf[o] = v;
          if (Cb) Cb[o] = (bf16_t)v;
        }
      }
}

// Fused pe4-generation + pos_fusion GEMM: rel_pos = relu(pe4 @ W + b), bf16.
// M=16384 (i*128+j), N=768, K=3072.  A tile synthesized in LDS from sinusoids.
__global__ __launch_bounds__(256) void k_posfusion(const bf16_t* __restrict__ W,
                                                   const float* __restrict__ bias,
                                                   bf16_t* __restrict__ Cb) {
  const int M = LLROWS, N = HH, K = 4 * HH;
  __shared__ __align__(16) bf16_t As[128][40];
  __shared__ __align__(16) bf16_t Bs[32][136];
  const int tid = threadIdx.x;
  const int wave = tid >> 5, lane = tid & 31;
  const int hsel = lane >> 4, l15 = lane & 15;
  const int m0 = blockIdx.y * 128, n0 = blockIdx.x * 128;
  const int wm = (wave & 3) * 32, wn = (wave >> 2) * 64;
  v8f acc[2][4];
#pragma unroll
  for (int i = 0; i < 2; ++i)
#pragma unroll
    for (int j = 0; j < 4; ++j) acc[i][j] = zero8();

  const int ar = tid >> 1, ac = (tid & 1) * 16;
  const int br = tid >> 3, bc = (tid & 7) * 16;
  // distances for this thread's A row:  m -> (i = m/128, j = m%128)
  const int m = m0 + ar;
  const int ri = m >> 7, rj = m & 127;
  const float psi = (float)(ri & 63), pei = (float)((ri & 63) + (ri >> 6));
  const float psj = (float)(rj & 63), pej = (float)((rj & 63) + (rj >> 6));
  float dist[4];
  dist[0] = psi - psj; dist[1] = psi - pej; dist[2] = pei - psj; dist[3] = pei - pej;

  for (int kt = 0; kt < K; kt += 32) {
    // synthesize pe4 A tile: k -> (blk = k/768, kk = k%768), half=384 sin|cos
#pragma unroll
    for (int e = 0; e < 16; ++e) {
      int k = kt + ac + e;
      int blk = k / HH;
      int kk = k - blk * HH;
      int f = (kk < 384) ? kk : kk - 384;
      float ang = dist[blk] * __expf(-0.0239852618f * (float)f);  // ln(1e4)/384
      float val = (kk < 384) ? __sinf(ang) : __cosf(ang);
      As[ar][ac + e] = (bf16_t)val;
    }
    {
      const bf16_t* s = W + (size_t)(kt + br) * N + n0 + bc;
      *(v8bf*)&Bs[br][bc] = *(const v8bf*)s;
      *(v8bf*)&Bs[br][bc + 8] = *(const v8bf*)(s + 8);
    }
    __syncthreads();
    v16bf bfr[4];
#pragma unroll
    for (int nf = 0; nf < 4; ++nf) {
#pragma unroll
      for (int s = 0; s < 16; ++s)
        bfr[nf][s] = Bs[hsel * 8 + (s & 7) + (s >> 3) * 16][wn + nf * 16 + l15];
    }
#pragma unroll
    for (int mf = 0; mf < 2; ++mf) {
      const int row = wm + mf * 16 + l15;
      v8bf a0 = *(const v8bf*)&As[row][hsel * 8];
      v8bf a1 = *(const v8bf*)&As[row][hsel * 8 + 16];
      v16bf af = mk16(a0, a1);
#pragma unroll
      for (int nf = 0; nf < 4; ++nf) acc[mf][nf] = wmma_bf16(af, bfr[nf], acc[mf][nf]);
    }
    __syncthreads();
  }
#pragma unroll
  for (int mf = 0; mf < 2; ++mf)
#pragma unroll
    for (int nf = 0; nf < 4; ++nf)
#pragma unroll
      for (int e = 0; e < 8; ++e) {
        int gr = m0 + wm + mf * 16 + e + 8 * hsel;
        int gc = n0 + wn + nf * 16 + l15;
        float v = fmaxf(acc[mf][nf][e] + bias[gc], 0.f);
        Cb[(size_t)gr * N + gc] = (bf16_t)v;
      }
  (void)M;
}

// score1[b,h,i,j] = (q+u)[b,h,i,:] . k[b,h,j,:]   (K^T via staging transpose)
__global__ __launch_bounds__(256) void k_score1(const bf16_t* __restrict__ qu,
                                                const bf16_t* __restrict__ kb,
                                                float* __restrict__ score) {
  const int bh = blockIdx.x;  // b*NH+h, 384 blocks
  const bf16_t* Ab = qu + (size_t)bh * LL * DHD;
  const bf16_t* Kb = kb + (size_t)bh * LL * DHD;
  float* S = score + (size_t)bh * LL * LL;
  __shared__ __align__(16) bf16_t As[128][40];
  __shared__ __align__(16) bf16_t Bs[32][136];
  const int tid = threadIdx.x;
  const int wave = tid >> 5, lane = tid & 31;
  const int hsel = lane >> 4, l15 = lane & 15;
  const int wm = (wave & 3) * 32, wn = (wave >> 2) * 64;
  v8f acc[2][4];
#pragma unroll
  for (int i = 0; i < 2; ++i)
#pragma unroll
    for (int j = 0; j < 4; ++j) acc[i][j] = zero8();

  const int ar = tid >> 1, ac = (tid & 1) * 16;
  const int tj = tid & 127, dh = (tid >> 7) * 16;
  for (int kt = 0; kt < DHD; kt += 32) {
    {
      const bf16_t* s = Ab + (size_t)ar * DHD + kt + ac;
      *(v8bf*)&As[ar][ac] = *(const v8bf*)s;
      *(v8bf*)&As[ar][ac + 8] = *(const v8bf*)(s + 8);
    }
    {  // transpose stage: Bs[d][j] = k[b,h,j,d]
      const bf16_t* s = Kb + (size_t)tj * DHD + kt + dh;
      v8bf k0 = *(const v8bf*)s;
      v8bf k1 = *(const v8bf*)(s + 8);
#pragma unroll
      for (int e = 0; e < 8; ++e) { Bs[dh + e][tj] = k0[e]; Bs[dh + 8 + e][tj] = k1[e]; }
    }
    __syncthreads();
    v16bf bfr[4];
#pragma unroll
    for (int nf = 0; nf < 4; ++nf) {
#pragma unroll
      for (int s = 0; s < 16; ++s)
        bfr[nf][s] = Bs[hsel * 8 + (s & 7) + (s >> 3) * 16][wn + nf * 16 + l15];
    }
#pragma unroll
    for (int mf = 0; mf < 2; ++mf) {
      const int row = wm + mf * 16 + l15;
      v8bf a0 = *(const v8bf*)&As[row][hsel * 8];
      v8bf a1 = *(const v8bf*)&As[row][hsel * 8 + 16];
      v16bf af = mk16(a0, a1);
#pragma unroll
      for (int nf = 0; nf < 4; ++nf) acc[mf][nf] = wmma_bf16(af, bfr[nf], acc[mf][nf]);
    }
    __syncthreads();
  }
#pragma unroll
  for (int mf = 0; mf < 2; ++mf)
#pragma unroll
    for (int nf = 0; nf < 4; ++nf)
#pragma unroll
      for (int e = 0; e < 8; ++e) {
        int gr = wm + mf * 16 + e + 8 * hsel;
        int gc = wn + nf * 16 + l15;
        S[(size_t)gr * LL + gc] = acc[mf][nf][e];
      }
}

// score2[b,h,i,j] += (q+vb)[b,h,i,:] . r[i,j,h,:]   block = (h, i); M=32 (b)
__global__ __launch_bounds__(256) void k_score2(const bf16_t* __restrict__ qv,
                                                const bf16_t* __restrict__ rmat,
                                                float* __restrict__ score) {
  const int h = blockIdx.x;  // 12
  const int i = blockIdx.y;  // 128
  __shared__ __align__(16) bf16_t As[32][72];
  __shared__ __align__(16) bf16_t Bs[64][136];
  const int tid = threadIdx.x;
  const int wave = tid >> 5, lane = tid & 31;
  const int hsel = lane >> 4, l15 = lane & 15;
  {  // stage A: [b][d] 32x64
    const int r = tid >> 3, c = (tid & 7) * 8;
    const bf16_t* s = qv + ((size_t)((r * NHD + h) * LL + i)) * DHD + c;
    *(v8bf*)&As[r][c] = *(const v8bf*)s;
  }
  {  // stage B^T: Bs[d][j] = r[i*128+j][h*64+d], 64x128
    const int tj = tid & 127, dh = (tid >> 7) * 32;
    const bf16_t* s = rmat + ((size_t)(i * LL + tj)) * HH + h * DHD + dh;
#pragma unroll
    for (int q4 = 0; q4 < 4; ++q4) {
      v8bf r0 = *(const v8bf*)(s + q4 * 8);
#pragma unroll
      for (int e = 0; e < 8; ++e) Bs[dh + q4 * 8 + e][tj] = r0[e];
    }
  }
  __syncthreads();
  const int mfh = wave & 1;        // 16-row half of the 32 batch rows
  const int nc = (wave >> 1) * 32; // 32-col slice of j
  v8f acc[2];
  acc[0] = zero8();
  acc[1] = zero8();
  for (int ks = 0; ks < DHD; ks += 32) {
    const int row = mfh * 16 + l15;
    v8bf a0 = *(const v8bf*)&As[row][ks + hsel * 8];
    v8bf a1 = *(const v8bf*)&As[row][ks + hsel * 8 + 16];
    v16bf af = mk16(a0, a1);
#pragma unroll
    for (int nf = 0; nf < 2; ++nf) {
      v16bf bf;
#pragma unroll
      for (int s = 0; s < 16; ++s)
        bf[s] = Bs[ks + hsel * 8 + (s & 7) + (s >> 3) * 16][nc + nf * 16 + l15];
      acc[nf] = wmma_bf16(af, bf, acc[nf]);
    }
  }
#pragma unroll
  for (int nf = 0; nf < 2; ++nf)
#pragma unroll
    for (int e = 0; e < 8; ++e) {
      int b = mfh * 16 + e + 8 * hsel;  // batch row
      int j = nc + nf * 16 + l15;
      size_t o = (((size_t)b * NHD + h) * LL + i) * LL + j;
      score[o] += acc[nf][e];  // each (h,i) block owns disjoint slice
    }
}

// ctx[b,i,h,:] = attn[b,h,i,:] @ v[b,h,:,:]   per (b,h): M=128 K=128 N=64
__global__ __launch_bounds__(256) void k_ctx(const bf16_t* __restrict__ attn,
                                             const bf16_t* __restrict__ vh,
                                             bf16_t* __restrict__ ctxb) {
  const int bh = blockIdx.x;
  const int b = bh / NHD, h = bh - b * NHD;
  const bf16_t* Ab = attn + (size_t)bh * LL * LL;
  const bf16_t* Vb = vh + (size_t)bh * LL * DHD;
  __shared__ __align__(16) bf16_t As[128][40];
  __shared__ __align__(16) bf16_t Bs[32][72];
  const int tid = threadIdx.x;
  const int wave = tid >> 5, lane = tid & 31;
  const int hsel = lane >> 4, l15 = lane & 15;
  const int wm = (wave & 3) * 32, wn = (wave >> 2) * 32;
  v8f acc[2][2];
#pragma unroll
  for (int i = 0; i < 2; ++i) { acc[i][0] = zero8(); acc[i][1] = zero8(); }

  const int ar = tid >> 1, ac = (tid & 1) * 16;
  const int br = tid >> 3, bc = (tid & 7) * 8;
  for (int kt = 0; kt < LL; kt += 32) {
    {
      const bf16_t* s = Ab + (size_t)ar * LL + kt + ac;
      *(v8bf*)&As[ar][ac] = *(const v8bf*)s;
      *(v8bf*)&As[ar][ac + 8] = *(const v8bf*)(s + 8);
    }
    *(v8bf*)&Bs[br][bc] = *(const v8bf*)(Vb + (size_t)(kt + br) * DHD + bc);
    __syncthreads();
    v16bf bfr[2];
#pragma unroll
    for (int nf = 0; nf < 2; ++nf) {
#pragma unroll
      for (int s = 0; s < 16; ++s)
        bfr[nf][s] = Bs[hsel * 8 + (s & 7) + (s >> 3) * 16][wn + nf * 16 + l15];
    }
#pragma unroll
    for (int mf = 0; mf < 2; ++mf) {
      const int row = wm + mf * 16 + l15;
      v8bf a0 = *(const v8bf*)&As[row][hsel * 8];
      v8bf a1 = *(const v8bf*)&As[row][hsel * 8 + 16];
      v16bf af = mk16(a0, a1);
#pragma unroll
      for (int nf = 0; nf < 2; ++nf) acc[mf][nf] = wmma_bf16(af, bfr[nf], acc[mf][nf]);
    }
    __syncthreads();
  }
#pragma unroll
  for (int mf = 0; mf < 2; ++mf)
#pragma unroll
    for (int nf = 0; nf < 2; ++nf)
#pragma unroll
      for (int e = 0; e < 8; ++e) {
        int gi = wm + mf * 16 + e + 8 * hsel;
        int gd = wn + nf * 16 + l15;
        ctxb[((size_t)(b * LL + gi)) * HH + h * DHD + gd] = (bf16_t)acc[mf][nf][e];
      }
}

// =========================== workspace layout ==============================
static constexpr size_t AL(size_t x) { return (x + 255) & ~(size_t)255; }
static constexpr size_t O_WSP = 0;
static constexpr size_t O_WRP = O_WSP + AL((size_t)INP * HH * 2);
static constexpr size_t O_WPF = O_WRP + AL((size_t)EPAD * HH * 2);
static constexpr size_t O_WQ  = O_WPF + AL((size_t)4 * HH * HH * 2);
static constexpr size_t O_WK  = O_WQ + AL((size_t)HH * HH * 2);
static constexpr size_t O_WV  = O_WK + AL((size_t)HH * HH * 2);
static constexpr size_t O_WR  = O_WV + AL((size_t)HH * HH * 2);
static constexpr size_t O_WO  = O_WR + AL((size_t)HH * HH * 2);
static constexpr size_t O_W1  = O_WO + AL((size_t)HH * HH * 2);
static constexpr size_t O_W2  = O_W1 + AL((size_t)HH * FFD * 2);
static constexpr size_t O_WFC = O_W2 + AL((size_t)FFD * HH * 2);
// scratch region (early buffers; attn bf16 reuses this region later)
static constexpr size_t O_SVB  = O_WFC + AL((size_t)HH * HQ * 2);
static constexpr size_t O_AREL = O_SVB + AL((size_t)BSROWS * INP * 2);
static constexpr size_t O_XS   = O_AREL + AL((size_t)BSROWS * EPAD * 2);
static constexpr size_t O_XR   = O_XS + AL((size_t)BSROWS * HH * 4);
static constexpr size_t O_ATTN = O_SVB;  // alias: 12.6MB fits in 18MB region
static constexpr size_t O_X    = O_XR + AL((size_t)BSROWS * HH * 4);
static constexpr size_t O_XB   = O_X + AL((size_t)BLROWS * HH * 4);
static constexpr size_t O_RELPOS = O_XB + AL((size_t)BLROWS * HH * 2);
static constexpr size_t O_SCORE  = O_RELPOS;  // alias (both 25,165,824 bytes)
static constexpr size_t O_R    = O_RELPOS + AL((size_t)LLROWS * HH * 2);
static constexpr size_t O_QF   = O_R + AL((size_t)LLROWS * HH * 2);
static constexpr size_t O_KF   = O_QF + AL((size_t)BLROWS * HH * 4);
static constexpr size_t O_VF   = O_KF + AL((size_t)BLROWS * HH * 4);
static constexpr size_t O_FB   = O_QF;  // alias: ffn hidden bf16 over q/k fp32
static constexpr size_t O_F2   = O_VF;  // alias: ffn out fp32 over v fp32
static constexpr size_t O_QU   = O_VF + AL((size_t)BLROWS * HH * 4);
static constexpr size_t O_QV   = O_QU + AL((size_t)BLROWS * HH * 2);
static constexpr size_t O_KB   = O_QV + AL((size_t)BLROWS * HH * 2);
static constexpr size_t O_VH   = O_KB + AL((size_t)BLROWS * HH * 2);
static constexpr size_t O_CTXB = O_VH + AL((size_t)BLROWS * HH * 2);
static constexpr size_t O_A    = O_CTXB + AL((size_t)BLROWS * HH * 2);
static constexpr size_t O_H1   = O_A + AL((size_t)BLROWS * HH * 4);
static constexpr size_t O_H1B  = O_H1 + AL((size_t)BLROWS * HH * 4);
static constexpr size_t O_H2B  = O_H1B + AL((size_t)BLROWS * HH * 2);
static constexpr size_t O_FCO  = O_H2B + AL((size_t)BLROWS * HH * 2);
static constexpr size_t O_NLL  = O_FCO + AL((size_t)BLROWS * HQ * 4);
static constexpr size_t O_VALID = O_NLL + AL((size_t)BLROWS * 4);

// =============================== driver ====================================
extern "C" void kernel_launch(void* const* d_in, const int* in_sizes, int n_in,
                              void* d_out, int out_size, void* d_ws, size_t ws_size,
                              hipStream_t stream) {
  // inputs in setup_inputs() dict order (params in source insertion order)
  const float* sent_vectors  = (const float*)d_in[0];
  const int*   sent_mask     = (const int*)d_in[1];
  const int*   sent_rel_ids  = (const int*)d_in[2];
  const int*   sent_rel_mask = (const int*)d_in[3];
  const int*   labels        = (const int*)d_in[4];
  const float* rel_embed     = (const float*)d_in[5];
  const float* sent_proj_w   = (const float*)d_in[6];
  const float* sent_proj_b   = (const float*)d_in[7];
  const float* rel_proj_w    = (const float*)d_in[8];
  const float* rel_proj_b    = (const float*)d_in[9];
  const float* pos_fusion_w  = (const float*)d_in[10];
  const float* pos_fusion_b  = (const float*)d_in[11];
  const float* wq = (const float*)d_in[12]; const float* bq = (const float*)d_in[13];
  const float* wk = (const float*)d_in[14]; const float* bk = (const float*)d_in[15];
  const float* wv = (const float*)d_in[16]; const float* bv = (const float*)d_in[17];
  const float* wr = (const float*)d_in[18];
  const float* wo = (const float*)d_in[19]; const float* bo = (const float*)d_in[20];
  const float* u_b  = (const float*)d_in[21];
  const float* vb_b = (const float*)d_in[22];
  const float* ln1_g = (const float*)d_in[23]; const float* ln1_b = (const float*)d_in[24];
  const float* ffn_w1 = (const float*)d_in[25]; const float* ffn_b1 = (const float*)d_in[26];
  const float* ffn_w2 = (const float*)d_in[27]; const float* ffn_b2 = (const float*)d_in[28];
  const float* ln2_g = (const float*)d_in[29]; const float* ln2_b = (const float*)d_in[30];
  const float* fc_w = (const float*)d_in[31]; const float* fc_b = (const float*)d_in[32];
  const float* cls_w = (const float*)d_in[33]; const float* cls_b = (const float*)d_in[34];

  char* w = (char*)d_ws;
  bf16_t* WSP = (bf16_t*)(w + O_WSP);   bf16_t* WRP = (bf16_t*)(w + O_WRP);
  bf16_t* WPF = (bf16_t*)(w + O_WPF);
  bf16_t* WQ = (bf16_t*)(w + O_WQ);     bf16_t* WK = (bf16_t*)(w + O_WK);
  bf16_t* WV = (bf16_t*)(w + O_WV);     bf16_t* WR = (bf16_t*)(w + O_WR);
  bf16_t* WO = (bf16_t*)(w + O_WO);     bf16_t* W1 = (bf16_t*)(w + O_W1);
  bf16_t* W2 = (bf16_t*)(w + O_W2);     bf16_t* WFC = (bf16_t*)(w + O_WFC);
  bf16_t* SVB = (bf16_t*)(w + O_SVB);   bf16_t* AREL = (bf16_t*)(w + O_AREL);
  float* XS = (float*)(w + O_XS);       float* XR = (float*)(w + O_XR);
  float* X = (float*)(w + O_X);         bf16_t* XB = (bf16_t*)(w + O_XB);
  bf16_t* RELPOS = (bf16_t*)(w + O_RELPOS);
  float* SCORE = (float*)(w + O_SCORE); bf16_t* RMAT = (bf16_t*)(w + O_R);
  float* QF = (float*)(w + O_QF);       float* KF = (float*)(w + O_KF);
  float* VF = (float*)(w + O_VF);
  bf16_t* QU = (bf16_t*)(w + O_QU);     bf16_t* QV = (bf16_t*)(w + O_QV);
  bf16_t* KB = (bf16_t*)(w + O_KB);     bf16_t* VH = (bf16_t*)(w + O_VH);
  bf16_t* ATTN = (bf16_t*)(w + O_ATTN); bf16_t* CTXB = (bf16_t*)(w + O_CTXB);
  float* AOUT = (float*)(w + O_A);      float* H1 = (float*)(w + O_H1);
  bf16_t* H1B = (bf16_t*)(w + O_H1B);   bf16_t* H2B = (bf16_t*)(w + O_H2B);
  bf16_t* FB = (bf16_t*)(w + O_FB);     float* F2 = (float*)(w + O_F2);
  float* FCO = (float*)(w + O_FCO);
  float* NLL = (float*)(w + O_NLL);     float* VALID = (float*)(w + O_VALID);
  float* OUT = (float*)d_out;

  auto cvt = [&](const float* s, bf16_t* d, size_t n) {
    k_f32_to_bf16<<<(unsigned)((n + 255) / 256), 256, 0, stream>>>(s, d, (unsigned)n);
  };
  auto gemm = [&](const bf16_t* A, const bf16_t* Bm, const float* bias, float* Cf,
                  bf16_t* Cb, int M, int N, int K, int relu) {
    dim3 g((N + 127) / 128, M / 128);
    k_gemm_bf16<<<g, 256, 0, stream>>>(A, Bm, bias, Cf, Cb, M, N, K, relu);
  };

  // ---- weight & activation conversion to bf16 ----
  cvt(sent_proj_w, WSP, (size_t)INP * HH);
  k_pad_rows_bf16<<<(EPAD * HH + 255) / 256, 256, 0, stream>>>(rel_proj_w, WRP, EE, EPAD, HH);
  cvt(pos_fusion_w, WPF, (size_t)4 * HH * HH);
  cvt(wq, WQ, (size_t)HH * HH); cvt(wk, WK, (size_t)HH * HH);
  cvt(wv, WV, (size_t)HH * HH); cvt(wr, WR, (size_t)HH * HH);
  cvt(wo, WO, (size_t)HH * HH);
  cvt(ffn_w1, W1, (size_t)HH * FFD);
  cvt(ffn_w2, W2, (size_t)FFD * HH);
  cvt(fc_w, WFC, (size_t)HH * HQ);
  cvt(sent_vectors, SVB, (size_t)BSROWS * INP);
  k_gather_rel<<<(BSROWS * EPAD + 255) / 256, 256, 0, stream>>>(rel_embed, sent_rel_ids, AREL);

  // ---- input projections, interleave into x ----
  gemm(SVB, WSP, sent_proj_b, XS, nullptr, BSROWS, HH, INP, 0);
  gemm(AREL, WRP, rel_proj_b, XR, nullptr, BSROWS, HH, EPAD, 0);
  k_interleave<<<(BLROWS * HH + 255) / 256, 256, 0, stream>>>(XS, XR, X, XB);

  // ---- fused pe4 + pos_fusion, then r = rel_pos @ wr ----
  {
    dim3 g(HH / 128, LLROWS / 128);
    k_posfusion<<<g, 256, 0, stream>>>(WPF, pos_fusion_b, RELPOS);
  }
  gemm(RELPOS, WR, nullptr, nullptr, RMAT, LLROWS, HH, HH, 0);

  // ---- q/k/v projections + bias prep ----
  gemm(XB, WQ, bq, QF, nullptr, BLROWS, HH, HH, 0);
  gemm(XB, WK, bk, KF, nullptr, BLROWS, HH, HH, 0);
  gemm(XB, WV, bv, VF, nullptr, BLROWS, HH, HH, 0);
  k_qkv_prep<<<(BB * NHD * LL * DHD + 255) / 256, 256, 0, stream>>>(QF, KF, VF, u_b, vb_b,
                                                                    QU, QV, KB, VH);

  // ---- attention scores, softmax, context ----
  k_score1<<<BB * NHD, 256, 0, stream>>>(QU, KB, SCORE);
  {
    dim3 g(NHD, LL);
    k_score2<<<g, 256, 0, stream>>>(QV, RMAT, SCORE);
  }
  k_softmax<<<(BB * NHD * LL) / 8, 256, 0, stream>>>(SCORE, ATTN, sent_mask, sent_rel_mask);
  k_ctx<<<BB * NHD, 256, 0, stream>>>(ATTN, VH, CTXB);

  // ---- output projection + LN + FFN + LN ----
  gemm(CTXB, WO, bo, AOUT, nullptr, BLROWS, HH, HH, 0);
  k_add_ln<<<BLROWS / 8, 256, 0, stream>>>(X, AOUT, ln1_g, ln1_b, H1, H1B);
  gemm(H1B, W1, ffn_b1, nullptr, FB, BLROWS, FFD, HH, 1);
  gemm(FB, W2, ffn_b2, F2, nullptr, BLROWS, HH, FFD, 0);
  k_add_ln<<<BLROWS / 8, 256, 0, stream>>>(H1, F2, ln2_g, ln2_b, nullptr, H2B);

  // ---- head: fc -> cls -> argmax / NLL -> loss ----
  gemm(H2B, WFC, fc_b, FCO, nullptr, BLROWS, HQ, HH, 0);
  k_cls<<<BLROWS / 256, 256, 0, stream>>>(FCO, cls_w, cls_b, labels, OUT, NLL, VALID);
  k_loss<<<1, 256, 0, stream>>>(NLL, VALID, OUT);

  (void)in_sizes; (void)n_in; (void)out_size; (void)ws_size;
}
